// PFNet_69329362092260
// MI455X (gfx1250) — matrix operand order repaired and checked by reading, not compiled, not run on hardware
//
#include <hip/hip_runtime.h>
#include <stdint.h>

// ---------------- problem constants ----------------
#define BSZ 4
#define NPT 4096
#define BN  (BSZ * NPT)       // 16384 nodes
#define ENCN 25
#define ENCP 32               // padded encoding width
#define H 256
#define DD 32
#define NCLS 8

typedef __bf16 bf16;
typedef __attribute__((ext_vector_type(8)))  bf16 bf16x8;
typedef __attribute__((ext_vector_type(16))) bf16 bf16x16;
typedef __attribute__((ext_vector_type(8)))  float v8f;
typedef __attribute__((ext_vector_type(4)))  int   v4i;

__device__ __forceinline__ float selu_f(float x) {
  const float sc = 1.0507009873554805f;
  const float sa = 1.7580993408473766f;   // scale * alpha
  return x > 0.f ? sc * x : sa * (__expf(x) - 1.f);
}

// A fragment (16x32 bf16): lane<16 holds K 0..15, lane>=16 K 16..31
__device__ __forceinline__ bf16x16 ld_fragA(const bf16* __restrict__ p) {
  p += (threadIdx.x & 16) >> 1;
  bf16x8 lo = *(const bf16x8*)p;
  bf16x8 hi = *(const bf16x8*)(p + 8);
  bf16x16 r;
#pragma unroll
  for (int i = 0; i < 8; ++i) { r[i] = lo[i]; r[i + 8] = hi[i]; }
  return r;
}

// ---------------- async global->LDS staging (gfx1250) ----------------
#if __has_builtin(__builtin_amdgcn_global_load_async_to_lds_b128)
#define HAVE_ASYNC_LDS 1
#else
#define HAVE_ASYNC_LDS 0
#endif

__device__ __forceinline__ void stage_b128(const bf16* __restrict__ g, bf16* l) {
#if HAVE_ASYNC_LDS
  __builtin_amdgcn_global_load_async_to_lds_b128(
      (__attribute__((address_space(1))) v4i*)(void*)g,
      (__attribute__((address_space(3))) v4i*)(void*)l, 0, 0);
#else
  *(bf16x8*)l = *(const bf16x8*)g;
#endif
}

template <int N>
__device__ __forceinline__ void wait_stage() {
#if HAVE_ASYNC_LDS
#if __has_builtin(__builtin_amdgcn_s_wait_asynccnt)
  __builtin_amdgcn_s_wait_asynccnt(N);
#else
  asm volatile("s_wait_asynccnt %0" ::"i"(N) : "memory");
#endif
#endif
}

// ---------------- encode: one-hot + props, selu, pad to 32 ----------------
__global__ void k_encode(const float* __restrict__ X, float* __restrict__ enc) {
  int idx = blockIdx.x * blockDim.x + threadIdx.x;
  if (idx >= BN * ENCP) return;
  int node = idx >> 5, c = idx & 31;
  float v = 0.f;
  if (c < 11) {
    int id = (int)X[node * 15];
    v = (c == id) ? 1.f : 0.f;
  } else if (c < 25) {
    v = X[node * 15 + (c - 10)];
  }
  enc[idx] = (c < 25) ? selu_f(v) : 0.f;
}

// ---------------- 25->256 layer (K too ragged for WMMA) ----------------
__global__ void k_gemm25(const float* __restrict__ enc, const float* __restrict__ W,
                         const float* __restrict__ bias, bf16* __restrict__ out) {
  __shared__ float se[ENCP];
  int node = blockIdx.x;
  int o = threadIdx.x;          // 0..255
  if (o < ENCP) se[o] = enc[node * ENCP + o];
  __syncthreads();
  float acc = bias[o];
#pragma unroll
  for (int k = 0; k < ENCN; ++k) acc += se[k] * W[k * H + o];
  out[(long long)node * H + o] = (bf16)selu_f(acc);
}

// ---------------- pack fp32 weight [Kin,Nin] -> WMMA B-fragment bf16 ----------------
__global__ void k_pack_w(const float* __restrict__ W, int Kin, int Nin,
                         int Kpad, int Npad, bf16* __restrict__ dst) {
  int total = (Kpad >> 5) * (Npad >> 4) * 512;
  int idx = blockIdx.x * blockDim.x + threadIdx.x;
  if (idx >= total) return;
  int j = idx & 15;
  int lane = (idx >> 4) & 31;
  int rest = idx >> 9;
  int NC = Npad >> 4;
  int nc = rest % NC, kc = rest / NC;
  int k = kc * 32 + ((lane >> 4) << 4) + j;
  int n = nc * 16 + (lane & 15);
  float v = (k < Kin && n < Nin) ? W[k * Nin + n] : 0.f;
  dst[idx] = (bf16)v;
}

// dc21|dc22 concat pack: K=256, N=64
__global__ void k_pack_dc2(const float* __restrict__ W1, const float* __restrict__ W2,
                           bf16* __restrict__ dst) {
  int idx = blockIdx.x * blockDim.x + threadIdx.x;
  if (idx >= 16384) return;
  int j = idx & 15;
  int lane = (idx >> 4) & 31;
  int rest = idx >> 9;
  int nc = rest & 3, kc = rest >> 2;
  int k = kc * 32 + ((lane >> 4) << 4) + j;
  int n = nc * 16 + (lane & 15);
  float v = (n < 32) ? W1[k * 32 + n] : W2[k * 32 + (n - 32)];
  dst[idx] = (bf16)v;
}

__global__ void k_bias_dc2(const float* __restrict__ b1, const float* __restrict__ b2,
                           float* __restrict__ dst) {
  int c = threadIdx.x;
  if (c < 64) dst[c] = (c < 32) ? b1[c] : b2[c - 32];
}

// ---------------- generic bf16 WMMA GEMM (software-pipelined) ----------------
// 8 waves; wave tile 16x64; block tile 128x64.
// mode 0: bf16 out; mode 1: bf16 + fp32 out; mode 2: out in packed-B layout
__global__ void k_gemm_wmma(const bf16* __restrict__ A, long long aBatch, int Kpad,
                            const bf16* __restrict__ Bpk, long long bBatch, int Npad,
                            const float* __restrict__ bias, int Nbias,
                            bf16* __restrict__ outB, float* __restrict__ outF,
                            long long outBatch, int ldo, int act, int mode) {
  int tid  = threadIdx.x;
  int wave = tid >> 5, lane = tid & 31;
  int lr = lane & 15, hi8 = (lane >> 4) * 8;
  int M0 = blockIdx.x * 128 + wave * 16;
  int N0 = blockIdx.y * 64;
  int batch = blockIdx.z;
  int NC = Npad >> 4;

  const bf16* Arow  = A + (long long)batch * aBatch + (long long)(M0 + lr) * Kpad;
  const bf16* Bbase = Bpk + (long long)batch * bBatch + ((long long)(N0 >> 4) << 9) + lane * 16;

  v8f z = {0.f, 0.f, 0.f, 0.f, 0.f, 0.f, 0.f, 0.f};
  v8f acc[4] = {z, z, z, z};

  // prologue loads for k0 = 0
  bf16x16 aN = ld_fragA(Arow + hi8);
  bf16x16 bN0 = *(const bf16x16*)(Bbase);
  bf16x16 bN1 = *(const bf16x16*)(Bbase + 512);
  bf16x16 bN2 = *(const bf16x16*)(Bbase + 1024);
  bf16x16 bN3 = *(const bf16x16*)(Bbase + 1536);

  int lastk = Kpad - 32;
  for (int k0 = 0; k0 < Kpad; k0 += 32) {
    bf16x16 aC = aN, bC0 = bN0, bC1 = bN1, bC2 = bN2, bC3 = bN3;
    if (k0 < lastk) {                 // prefetch next iteration's fragments
      const bf16* bp = Bbase + (((long long)((k0 >> 5) + 1) * NC) << 9);
      aN  = ld_fragA(Arow + k0 + 32 + hi8);
      bN0 = *(const bf16x16*)(bp);
      bN1 = *(const bf16x16*)(bp + 512);
      bN2 = *(const bf16x16*)(bp + 1024);
      bN3 = *(const bf16x16*)(bp + 1536);
      __builtin_prefetch(Arow + k0 + 160, 0, 1);
    }
    acc[0] = __builtin_amdgcn_wmma_f32_16x16x32_bf16(false, aC, false, bC0, (short)0, acc[0], false, false);
    acc[1] = __builtin_amdgcn_wmma_f32_16x16x32_bf16(false, aC, false, bC1, (short)0, acc[1], false, false);
    acc[2] = __builtin_amdgcn_wmma_f32_16x16x32_bf16(false, aC, false, bC2, (short)0, acc[2], false, false);
    acc[3] = __builtin_amdgcn_wmma_f32_16x16x32_bf16(false, aC, false, bC3, (short)0, acc[3], false, false);
  }

#pragma unroll
  for (int nc = 0; nc < 4; ++nc) {
    int c = N0 + nc * 16 + lr;
    float bv = (bias != nullptr && c < Nbias) ? bias[c] : 0.f;
#pragma unroll
    for (int v = 0; v < 8; ++v) {
      int r = M0 + v + hi8;
      float x = acc[nc][v] + bv;
      if (act) x = selu_f(x);
      if (mode == 2) {
        int m = r & (NPT - 1);
        int bt = r >> 12;
        long long idx = (long long)bt * 1048576LL
                      + ((((long long)(m >> 5)) * NC + (c >> 4)) << 9)
                      + (((c & 15) + ((m >> 4) & 1) * 16) << 4) + (m & 15);
        outB[idx] = (bf16)x;
      } else {
        long long oi = (long long)batch * outBatch + (long long)r * ldo + c;
        outB[oi] = (bf16)x;
        if (mode == 1) outF[oi] = x;
      }
    }
  }
}

// ---------------- conv GEMM: selu(adj4 @ xW + b), K=4096, N=256, LDS-staged B ----------
// 8 waves; wave tile 32x64 (2 A frags x 4 B frags = 8 WMMA / K-step); block tile 256x64.
// B panel (4KB per K-step) staged via GLOBAL_LOAD_ASYNC_TO_LDS_B128, double buffered.
__global__ void k_gemm_conv(const bf16* __restrict__ A, const bf16* __restrict__ Bpk,
                            const float* __restrict__ bias, bf16* __restrict__ outB) {
  __shared__ bf16 sB[2][2048];
  int tid = threadIdx.x;
  int wave = tid >> 5, lane = tid & 31;
  int lr = lane & 15, hi8 = (lane >> 4) * 8;
  int M0 = blockIdx.x * 256 + wave * 32;
  int N0 = blockIdx.y * 64;
  int batch = blockIdx.z;

  const bf16* Arow0 = A + (long long)batch * NPT * NPT + (long long)(M0 + lr) * NPT;
  const bf16* Arow1 = Arow0 + 16LL * NPT;
  // contiguous 2048-element B panel per K-step: (kc*16 + N0/16)*512
  const bf16* Bpanel = Bpk + (long long)batch * 1048576LL + ((long long)(N0 >> 4) << 9);

  v8f z = {0.f, 0.f, 0.f, 0.f, 0.f, 0.f, 0.f, 0.f};
  v8f acc0[4] = {z, z, z, z};
  v8f acc1[4] = {z, z, z, z};

  // stage k-step 0
  stage_b128(Bpanel + tid * 8, &sB[0][tid * 8]);
  bf16x16 aN0 = ld_fragA(Arow0 + hi8);
  bf16x16 aN1 = ld_fragA(Arow1 + hi8);

  const int KC = NPT / 32;   // 128
  for (int kc = 0; kc < KC; ++kc) {
    int cur = kc & 1;
    bf16x16 aC0 = aN0, aC1 = aN1;
    if (kc + 1 < KC) {
      stage_b128(Bpanel + ((long long)(kc + 1) << 13) + tid * 8, &sB[cur ^ 1][tid * 8]);
      wait_stage<1>();                           // stage kc complete (kc+1 in flight)
      int k0 = (kc + 1) * 32;
      aN0 = ld_fragA(Arow0 + k0 + hi8);
      aN1 = ld_fragA(Arow1 + k0 + hi8);
      __builtin_prefetch(Arow0 + k0 + 128, 0, 1);
      __builtin_prefetch(Arow1 + k0 + 128, 0, 1);
    } else {
      wait_stage<0>();
    }
    __syncthreads();                             // sB[cur] visible to all waves
#pragma unroll
    for (int nc = 0; nc < 4; ++nc) {
      bf16x16 b = *(const bf16x16*)(&sB[cur][(nc << 9) + lane * 16]);
      acc0[nc] = __builtin_amdgcn_wmma_f32_16x16x32_bf16(false, aC0, false, b, (short)0, acc0[nc], false, false);
      acc1[nc] = __builtin_amdgcn_wmma_f32_16x16x32_bf16(false, aC1, false, b, (short)0, acc1[nc], false, false);
    }
    __syncthreads();                             // all waves done reading sB[cur]
  }

  long long obase = (long long)batch * NPT * H;
#pragma unroll
  for (int nc = 0; nc < 4; ++nc) {
    int c = N0 + nc * 16 + lr;
    float bv = bias[c];
#pragma unroll
    for (int v = 0; v < 8; ++v) {
      int r0 = M0 + v + hi8;
      outB[obase + (long long)r0 * H + c]        = (bf16)selu_f(acc0[nc][v] + bv);
      outB[obase + (long long)(r0 + 16) * H + c] = (bf16)selu_f(acc1[nc][v] + bv);
    }
  }
}

// ---------------- norms of c1/c2 ----------------
__global__ void k_norms(const float* __restrict__ c12, float* __restrict__ na,
                        float* __restrict__ nb) {
  int node = blockIdx.x * blockDim.x + threadIdx.x;
  if (node >= BN) return;
  const float* r = c12 + (long long)node * 64;
  float s1 = 0.f, s2 = 0.f;
#pragma unroll
  for (int i = 0; i < 32; ++i) { s1 += r[i] * r[i]; s2 += r[32 + i] * r[32 + i]; }
  na[node] = s1;
  nb[node] = s2;
}

// ---------------- pairwise adj^4 = exp(-4*sqrt(clip(Dsq))) via WMMA ----------------
__global__ void k_pairwise(const bf16* __restrict__ c12, const float* __restrict__ na,
                           const float* __restrict__ nb, bf16* __restrict__ adj4) {
  int tid = threadIdx.x;
  int wave = tid >> 5, lane = tid & 31;
  int lr = lane & 15, hi8 = (lane >> 4) * 8;
  int n0 = blockIdx.x * 128 + wave * 16;
  int m0 = blockIdx.y * 64;
  int batch = blockIdx.z;
  long long rb = (long long)batch * NPT;

  bf16x16 a = ld_fragA(c12 + (rb + n0 + lr) * 64 + hi8);

  v8f z = {0.f, 0.f, 0.f, 0.f, 0.f, 0.f, 0.f, 0.f};
  v8f acc[4] = {z, z, z, z};
  int koff = (lane >> 4) * 16;
#pragma unroll
  for (int nc = 0; nc < 4; ++nc) {
    bf16x16 b = *(const bf16x16*)(c12 + (rb + m0 + nc * 16 + lr) * 64 + 32 + koff);
    acc[nc] = __builtin_amdgcn_wmma_f32_16x16x32_bf16(false, a, false, b, (short)0, acc[nc], false, false);
  }

  long long obase = (long long)batch * NPT * NPT;
#pragma unroll
  for (int nc = 0; nc < 4; ++nc) {
    int m = m0 + nc * 16 + lr;
    float nbv = nb[rb + m];
#pragma unroll
    for (int v = 0; v < 8; ++v) {
      int n = n0 + v + hi8;
      float d = na[rb + n] - 2.f * acc[nc][v] + nbv;
      d = fminf(fmaxf(d, 1e-12f), 1e12f);
      float val = __expf(-4.f * sqrtf(d));
      adj4[obase + (long long)n * NPT + m] = (bf16)val;
    }
  }
}

// ---------------- small output heads: oid (8) + och (2) ----------------
__global__ void k_heads(const bf16* __restrict__ a3,
                        const float* __restrict__ Wo, const float* __restrict__ bo,
                        const float* __restrict__ Wc, const float* __restrict__ bc,
                        float* __restrict__ logits, float* __restrict__ charge) {
  int node = blockIdx.x * blockDim.x + threadIdx.x;
  if (node >= BN) return;
  float ao[8], ac[2];
#pragma unroll
  for (int o = 0; o < 8; ++o) ao[o] = bo[o];
  ac[0] = bc[0]; ac[1] = bc[1];
  const bf16* row = a3 + (long long)node * H;
  for (int k = 0; k < H; ++k) {
    float v = (float)row[k];
#pragma unroll
    for (int o = 0; o < 8; ++o) ao[o] += v * Wo[k * 8 + o];
    ac[0] += v * Wc[k * 2];
    ac[1] += v * Wc[k * 2 + 1];
  }
#pragma unroll
  for (int o = 0; o < 8; ++o) logits[node * 8 + o] = ao[o];
  charge[node * 2]     = ac[0];
  charge[node * 2 + 1] = ac[1];
}

// ---------------- build xb = [xconv | selu(logits) | zero-pad] width 288 ----------------
__global__ void k_xb(const bf16* __restrict__ xconv, const float* __restrict__ logits,
                     bf16* __restrict__ xb) {
  int idx = blockIdx.x * blockDim.x + threadIdx.x;
  if (idx >= BN * 288) return;
  int node = idx / 288, c = idx - node * 288;
  float v = 0.f;
  if (c < 256)      v = (float)xconv[(long long)node * H + c];
  else if (c < 264) v = selu_f(logits[node * 8 + (c - 256)]);
  xb[idx] = (bf16)v;
}

// ---------------- momentum head output (3) ----------------
__global__ void k_om(const bf16* __restrict__ bb3, const float* __restrict__ W,
                     const float* __restrict__ b, float* __restrict__ corr) {
  int node = blockIdx.x * blockDim.x + threadIdx.x;
  if (node >= BN) return;
  float a0 = b[0], a1 = b[1], a2 = b[2];
  const bf16* row = bb3 + (long long)node * H;
  for (int k = 0; k < H; ++k) {
    float v = (float)row[k];
    a0 += v * W[k * 3];
    a1 += v * W[k * 3 + 1];
    a2 += v * W[k * 3 + 2];
  }
  corr[node * 3]     = a0;
  corr[node * 3 + 1] = a1;
  corr[node * 3 + 2] = a2;
}

// ---------------- final assembly [BN,13] ----------------
__global__ void k_final(const float* __restrict__ X, const float* __restrict__ logits,
                        const float* __restrict__ corr, const float* __restrict__ charge,
                        float* __restrict__ out) {
  int idx = blockIdx.x * blockDim.x + threadIdx.x;
  if (idx >= BN * 13) return;
  int node = idx / 13, c = idx - node * 13;
  float v;
  if (c < 8)       v = logits[node * 8 + c];
  else if (c < 11) v = X[node * 15 + 2 + (c - 8)] + corr[node * 3 + (c - 8)];
  else             v = charge[node * 2 + (c - 11)];
  out[idx] = v;
}

// ---------------- host ----------------
static inline char* take(char*& p, size_t bytes) {
  char* r = p;
  p += (bytes + 255) & ~size_t(255);
  return r;
}

extern "C" void kernel_launch(void* const* d_in, const int* in_sizes, int n_in,
                              void* d_out, int out_size, void* d_ws, size_t ws_size,
                              hipStream_t stream) {
  (void)in_sizes; (void)n_in; (void)out_size; (void)ws_size;
  const float* X      = (const float*)d_in[0];
  const float* W_dc1  = (const float*)d_in[1];  const float* b_dc1  = (const float*)d_in[2];
  const float* W_dc21 = (const float*)d_in[3];  const float* b_dc21 = (const float*)d_in[4];
  const float* W_dc22 = (const float*)d_in[5];  const float* b_dc22 = (const float*)d_in[6];
  const float* W_in1  = (const float*)d_in[7];  const float* b_in1  = (const float*)d_in[8];
  const float* W_in2  = (const float*)d_in[9];  const float* b_in2  = (const float*)d_in[10];
  const float* W_in3  = (const float*)d_in[11]; const float* b_in3  = (const float*)d_in[12];
  const float* W_conv = (const float*)d_in[13]; const float* b_conv = (const float*)d_in[14];
  const float* W_id1  = (const float*)d_in[15]; const float* b_id1  = (const float*)d_in[16];
  const float* W_id2  = (const float*)d_in[17]; const float* b_id2  = (const float*)d_in[18];
  const float* W_id3  = (const float*)d_in[19]; const float* b_id3  = (const float*)d_in[20];
  const float* W_oid  = (const float*)d_in[21]; const float* b_oid  = (const float*)d_in[22];
  const float* W_och  = (const float*)d_in[23]; const float* b_och  = (const float*)d_in[24];
  const float* W_m1   = (const float*)d_in[25]; const float* b_m1   = (const float*)d_in[26];
  const float* W_m2   = (const float*)d_in[27]; const float* b_m2   = (const float*)d_in[28];
  const float* W_m3   = (const float*)d_in[29]; const float* b_m3   = (const float*)d_in[30];
  const float* W_om   = (const float*)d_in[31]; const float* b_om   = (const float*)d_in[32];
  float* out = (float*)d_out;

  char* p = (char*)d_ws;
  float* enc   = (float*)take(p, (size_t)BN * ENCP * 4);
  bf16*  h_dc  = (bf16*) take(p, (size_t)BN * H * 2);
  bf16*  x1    = (bf16*) take(p, (size_t)BN * H * 2);
  bf16*  x2    = (bf16*) take(p, (size_t)BN * H * 2);
  bf16*  x3    = (bf16*) take(p, (size_t)BN * H * 2);
  float* c12f  = (float*)take(p, (size_t)BN * 64 * 4);
  bf16*  c12b  = (bf16*) take(p, (size_t)BN * 64 * 2);
  float* na    = (float*)take(p, (size_t)BN * 4);
  float* nb    = (float*)take(p, (size_t)BN * 4);
  bf16*  adj4  = (bf16*) take(p, (size_t)BSZ * NPT * NPT * 2);
  bf16*  xWpk  = (bf16*) take(p, (size_t)BSZ * 1048576 * 2);
  bf16*  xconv = (bf16*) take(p, (size_t)BN * H * 2);
  bf16*  a1    = (bf16*) take(p, (size_t)BN * H * 2);
  bf16*  a2    = (bf16*) take(p, (size_t)BN * H * 2);
  bf16*  a3    = (bf16*) take(p, (size_t)BN * H * 2);
  float* logit = (float*)take(p, (size_t)BN * 8 * 4);
  float* chrg  = (float*)take(p, (size_t)BN * 2 * 4);
  bf16*  xb    = (bf16*) take(p, (size_t)BN * 288 * 2);
  bf16*  bb1   = (bf16*) take(p, (size_t)BN * 320 * 2);
  bf16*  bb2   = (bf16*) take(p, (size_t)BN * H * 2);
  bf16*  bb3   = (bf16*) take(p, (size_t)BN * H * 2);
  float* corr  = (float*)take(p, (size_t)BN * 3 * 4);
  bf16*  dc2pk = (bf16*) take(p, (size_t)16384 * 2);
  float* dc2b  = (float*)take(p, (size_t)64 * 4);
  bf16*  in2pk = (bf16*) take(p, (size_t)65536 * 2);
  bf16*  in3pk = (bf16*) take(p, (size_t)65536 * 2);
  bf16*  cvpk  = (bf16*) take(p, (size_t)65536 * 2);
  bf16*  id1pk = (bf16*) take(p, (size_t)65536 * 2);
  bf16*  id2pk = (bf16*) take(p, (size_t)65536 * 2);
  bf16*  id3pk = (bf16*) take(p, (size_t)65536 * 2);
  bf16*  m1pk  = (bf16*) take(p, (size_t)92160 * 2);
  bf16*  m2pk  = (bf16*) take(p, (size_t)81920 * 2);
  bf16*  m3pk  = (bf16*) take(p, (size_t)65536 * 2);

  // --- weight packing ---
  k_pack_dc2<<<64, 256, 0, stream>>>(W_dc21, W_dc22, dc2pk);
  k_bias_dc2<<<1, 64, 0, stream>>>(b_dc21, b_dc22, dc2b);
  k_pack_w<<<256, 256, 0, stream>>>(W_in2, 256, 256, 256, 256, in2pk);
  k_pack_w<<<256, 256, 0, stream>>>(W_in3, 256, 256, 256, 256, in3pk);
  k_pack_w<<<256, 256, 0, stream>>>(W_conv, 256, 256, 256, 256, cvpk);
  k_pack_w<<<256, 256, 0, stream>>>(W_id1, 256, 256, 256, 256, id1pk);
  k_pack_w<<<256, 256, 0, stream>>>(W_id2, 256, 256, 256, 256, id2pk);
  k_pack_w<<<256, 256, 0, stream>>>(W_id3, 256, 256, 256, 256, id3pk);
  k_pack_w<<<360, 256, 0, stream>>>(W_m1, 264, 264, 288, 320, m1pk);
  k_pack_w<<<320, 256, 0, stream>>>(W_m2, 264, 256, 320, 256, m2pk);
  k_pack_w<<<256, 256, 0, stream>>>(W_m3, 256, 256, 256, 256, m3pk);

  // --- encode + 25->256 layers ---
  k_encode<<<(BN * ENCP + 255) / 256, 256, 0, stream>>>(X, enc);
  k_gemm25<<<BN, 256, 0, stream>>>(enc, W_dc1, b_dc1, h_dc);
  k_gemm25<<<BN, 256, 0, stream>>>(enc, W_in1, b_in1, x1);

  // --- distance coords: c1|c2 = h_dc @ [W_dc21|W_dc22] ---
  k_gemm_wmma<<<dim3(128, 1, 1), 256, 0, stream>>>(
      h_dc, 0, 256, dc2pk, 0, 64, dc2b, 64, c12b, c12f, 0, 64, 0, 1);
  k_norms<<<(BN + 255) / 256, 256, 0, stream>>>(c12f, na, nb);

  // --- adjacency^4 (bf16, fused exp(-4*sqrt)) ---
  k_pairwise<<<dim3(NPT / 128, NPT / 64, BSZ), 256, 0, stream>>>(c12b, na, nb, adj4);

  // --- node MLP ---
  k_gemm_wmma<<<dim3(128, 4, 1), 256, 0, stream>>>(
      x1, 0, 256, in2pk, 0, 256, b_in2, 256, x2, nullptr, 0, 256, 1, 0);
  k_gemm_wmma<<<dim3(128, 4, 1), 256, 0, stream>>>(
      x2, 0, 256, in3pk, 0, 256, b_in3, 256, x3, nullptr, 0, 256, 1, 0);

  // --- xW = x3 @ W_conv, stored directly in packed-B layout per batch ---
  k_gemm_wmma<<<dim3(128, 4, 1), 256, 0, stream>>>(
      x3, 0, 256, cvpk, 0, 256, nullptr, 0, xWpk, nullptr, 0, 256, 0, 2);

  // --- conv: selu(adj4 @ xW + b_conv), batched, LDS-staged B, 32x64 wave tile ---
  k_gemm_conv<<<dim3(NPT / 256, 4, BSZ), 256, 0, stream>>>(adj4, xWpk, b_conv, xconv);

  // --- id head ---
  k_gemm_wmma<<<dim3(128, 4, 1), 256, 0, stream>>>(
      xconv, 0, 256, id1pk, 0, 256, b_id1, 256, a1, nullptr, 0, 256, 1, 0);
  k_gemm_wmma<<<dim3(128, 4, 1), 256, 0, stream>>>(
      a1, 0, 256, id2pk, 0, 256, b_id2, 256, a2, nullptr, 0, 256, 1, 0);
  k_gemm_wmma<<<dim3(128, 4, 1), 256, 0, stream>>>(
      a2, 0, 256, id3pk, 0, 256, b_id3, 256, a3, nullptr, 0, 256, 1, 0);
  k_heads<<<(BN + 255) / 256, 256, 0, stream>>>(a3, W_oid, b_oid, W_och, b_och, logit, chrg);

  // --- momentum head ---
  k_xb<<<(BN * 288 + 255) / 256, 256, 0, stream>>>(xconv, logit, xb);
  k_gemm_wmma<<<dim3(128, 5, 1), 256, 0, stream>>>(
      xb, 0, 288, m1pk, 0, 320, b_m1, 264, bb1, nullptr, 0, 320, 1, 0);
  k_gemm_wmma<<<dim3(128, 4, 1), 256, 0, stream>>>(
      bb1, 0, 320, m2pk, 0, 256, b_m2, 256, bb2, nullptr, 0, 256, 1, 0);
  k_gemm_wmma<<<dim3(128, 4, 1), 256, 0, stream>>>(
      bb2, 0, 256, m3pk, 0, 256, b_m3, 256, bb3, nullptr, 0, 256, 1, 0);
  k_om<<<(BN + 255) / 256, 256, 0, stream>>>(bb3, W_om, b_om, corr);

  // --- final output [BN,13] ---
  k_final<<<(BN * 13 + 255) / 256, 256, 0, stream>>>(X, logit, corr, chrg, out);
}
